// GATRegressor_26431228740369
// MI455X (gfx1250) — matrix operand order repaired and checked by reading, not compiled, not run on hardware
//
#include <hip/hip_runtime.h>
#include <hip/hip_bf16.h>

// ---------------------------------------------------------------------------
// GAT regressor for MI455X (gfx1250, wave32).
// Dense projections: V_WMMA_F32_16X16X32_BF16, A via b128 vector loads,
// B pre-packed into WMMA lane layout (32 contiguous bytes per lane fragment).
// Edge softmax/scatter: f32 global atomics (working set is L2-resident).
// ---------------------------------------------------------------------------

typedef __attribute__((ext_vector_type(16))) __bf16 v16bf;
typedef __attribute__((ext_vector_type(8)))  __bf16 v8bf;
typedef __attribute__((ext_vector_type(8)))  float  v8f;

#define NEG_SLOPE 0.2f

// ---------------------------- utility kernels ------------------------------

__global__ void k_cvt_bf16(const float* __restrict__ in, __bf16* __restrict__ out,
                           long long n) {
  long long i = (blockIdx.x * (long long)blockDim.x + threadIdx.x) * 4;
  if (i + 3 < n) {
    const float4 v = *(const float4*)(in + i);
    out[i]     = (__bf16)v.x;
    out[i + 1] = (__bf16)v.y;
    out[i + 2] = (__bf16)v.z;
    out[i + 3] = (__bf16)v.w;
  } else {
    for (; i < n; ++i) out[i] = (__bf16)in[i];
  }
}

__global__ void k_fill_f32(float* __restrict__ p, float v, long long n) {
  long long i = blockIdx.x * (long long)blockDim.x + threadIdx.x;
  if (i < n) p[i] = v;
}

__global__ void k_fill_u32(unsigned* __restrict__ p, unsigned v, long long n) {
  long long i = blockIdx.x * (long long)blockDim.x + threadIdx.x;
  if (i < n) p[i] = v;
}

// --------------------- B-matrix packing for WMMA ---------------------------
// Pack f32 B[K,N] row-major into the wave32 B-fragment layout so each lane's
// 16-bf16 fragment is contiguous (32 bytes -> 2x global_load_b128):
//   Bp[((k0/32)*(N/16) + nt)*512 + lane*16 + t] = bf16(B[k, n])
//   with n = nt*16 + (lane&15), k = k0 + (t>>3)*16 + (lane>>4)*8 + (t&7)
__global__ void k_pack_b(const float* __restrict__ B, __bf16* __restrict__ Bp,
                         int N, int K) {
  const long long i = blockIdx.x * (long long)blockDim.x + threadIdx.x;
  const long long total = (long long)K * N;
  if (i >= total) return;
  const int t    = (int)(i & 15);
  const int lane = (int)((i >> 4) & 31);
  const long long rest = i >> 9;
  const int ntiles = N >> 4;
  const int nt = (int)(rest % ntiles);
  const int kb = (int)(rest / ntiles);
  const int n = nt * 16 + (lane & 15);
  const int k = kb * 32 + (t >> 3) * 16 + (lane >> 4) * 8 + (t & 7);
  Bp[i] = (__bf16)B[(size_t)k * N + n];
}

// --------------------------- WMMA bf16 GEMM --------------------------------
// C[M,N] = A[M,K] * Bp(packed), f32 accumulate. One wave per 16 x (16*NT)
// output tile; the A fragment is reused across NT accumulators.
// Requires M % 16 == 0, K % 32 == 0, N % (16*NT) == 0.
// Layouts per CDNA5 ISA 7.12.2 (wave32):
//   A 16x32 bf16: lane {m=l&15, khalf=l>>4}; group g: 8 contiguous bf16 at
//                 k = k0 + g*16 + khalf*8  (16B-aligned -> b128 load)
//   C/D 16x16 f32: elem v -> (m = (l>>4)*8 + v, n = l&15)
template <int NT>
__global__ void k_wmma_gemm_bf16(const __bf16* __restrict__ A,
                                 const __bf16* __restrict__ Bp,
                                 float* __restrict__ C,
                                 int M, int N, int K) {
  const int lane  = threadIdx.x & 31;
  const int lm    = lane & 15;
  const int khalf = lane >> 4;
  const int m0    = blockIdx.x * 16;
  const int nt0   = blockIdx.y * NT;
  const int ntiles = N >> 4;

  const __bf16* __restrict__ Arow = A + (size_t)(m0 + lm) * K;

  v8f acc[NT];
#pragma unroll
  for (int j = 0; j < NT; ++j) acc[j] = (v8f){};

  for (int k0 = 0; k0 < K; k0 += 32) {
    const v8bf alo = *(const v8bf*)(Arow + k0 + khalf * 8);        // k0..(+16)
    const v8bf ahi = *(const v8bf*)(Arow + k0 + 16 + khalf * 8);   // k0+16..
    const v16bf a = __builtin_shufflevector(alo, ahi,
        0, 1, 2, 3, 4, 5, 6, 7, 8, 9, 10, 11, 12, 13, 14, 15);

    const __bf16* __restrict__ bbase =
        Bp + (((size_t)(k0 >> 5) * ntiles + nt0) << 9) + lane * 16;
#pragma unroll
    for (int j = 0; j < NT; ++j) {
      const v16bf b = *(const v16bf*)(bbase + (j << 9));           // 2x b128
      acc[j] = __builtin_amdgcn_wmma_f32_16x16x32_bf16(
          /*neg_a=*/false, a, /*neg_b=*/false, b,
          /*c_mod=*/(short)0, acc[j], /*reuse_a=*/false, /*reuse_b=*/false);
    }
  }

#pragma unroll
  for (int j = 0; j < NT; ++j) {
    float* __restrict__ col = C + (size_t)(m0 + khalf * 8) * N + (nt0 + j) * 16 + lm;
#pragma unroll
    for (int v = 0; v < 8; ++v) col[(size_t)v * N] = acc[j][v];
  }
}

// ----------------------- attention coefficient dots ------------------------
// One wave per node: as[n] = h[n].a_src ; ad[n] = h[n].a_dst
__global__ void k_alpha(const float* __restrict__ h,
                        const float* __restrict__ a_s,
                        const float* __restrict__ a_d,
                        float* __restrict__ out_s, float* __restrict__ out_d,
                        int N, int F) {
  const long long gtid = blockIdx.x * (long long)blockDim.x + threadIdx.x;
  const long long node = gtid >> 5;
  const int lane = threadIdx.x & 31;
  if (node >= N) return;
  const float* __restrict__ row = h + (size_t)node * F;
  float ss = 0.f, sd = 0.f;
  for (int f = lane; f < F; f += 32) {
    const float v = row[f];
    ss += v * a_s[f];
    sd += v * a_d[f];
  }
#pragma unroll
  for (int off = 16; off; off >>= 1) {
    ss += __shfl_xor(ss, off, 32);
    sd += __shfl_xor(sd, off, 32);
  }
  if (lane == 0) { out_s[node] = ss; out_d[node] = sd; }
}

// --------------------------- edge-pass kernels -----------------------------
// Monotonic uint encoding of f32 so unsigned atomicMax == float max.
__device__ __forceinline__ unsigned f2key(float f) {
  const unsigned u = __float_as_uint(f);
  return (u & 0x80000000u) ? ~u : (u | 0x80000000u);
}
__device__ __forceinline__ float key2f(unsigned k) {
  const unsigned u = (k & 0x80000000u) ? (k & 0x7FFFFFFFu) : ~k;
  return __uint_as_float(u);
}

// Edges 0..E-1 come from edge_index; edges E..ET-1 are self-loops (i,i).
__device__ __forceinline__ void edge_sd(const long long* __restrict__ ei,
                                        long long E, long long i,
                                        int& s, int& d) {
  if (i < E) { s = (int)ei[i]; d = (int)ei[E + i]; }
  else       { s = d = (int)(i - E); }
}

// pass 1: e = leakyrelu(as[src] + ad[dst]); segment-max into mkey[dst]
__global__ void k_edge_lrelu_max(const long long* __restrict__ ei,
                                 long long E, long long ET,
                                 const float* __restrict__ as_,
                                 const float* __restrict__ ad_,
                                 float* __restrict__ ebuf,
                                 unsigned* __restrict__ mkey) {
  const long long i = blockIdx.x * (long long)blockDim.x + threadIdx.x;
  if (i >= ET) return;
  int s, d; edge_sd(ei, E, i, s, d);
  float v = as_[s] + ad_[d];
  v = v > 0.f ? v : NEG_SLOPE * v;
  ebuf[i] = v;
  atomicMax(mkey + d, f2key(v));
}

// pass 2: ex = exp(e - m[dst]); segment-sum into denom[dst]
__global__ void k_edge_exp_sum(const long long* __restrict__ ei,
                               long long E, long long ET,
                               const float* __restrict__ ebuf,
                               const unsigned* __restrict__ mkey,
                               float* __restrict__ exbuf,
                               float* __restrict__ denom) {
  const long long i = blockIdx.x * (long long)blockDim.x + threadIdx.x;
  if (i >= ET) return;
  int s, d; edge_sd(ei, E, i, s, d);
  const float ex = __expf(ebuf[i] - key2f(mkey[d]));
  exbuf[i] = ex;
  atomicAdd(denom + d, ex);
}

// pass 3: out[dst] += (ex/denom[dst]) * h[src]   (one wave per edge)
__global__ void k_edge_agg(const long long* __restrict__ ei,
                           long long E, long long ET,
                           const float* __restrict__ h,
                           const float* __restrict__ exbuf,
                           const float* __restrict__ denom,
                           float* __restrict__ out, int F) {
  const long long gtid = blockIdx.x * (long long)blockDim.x + threadIdx.x;
  const long long i = gtid >> 5;
  const int lane = threadIdx.x & 31;
  if (i >= ET) return;
  int s, d; edge_sd(ei, E, i, s, d);
  const float alpha = exbuf[i] / denom[d];
  const float* __restrict__ hs = h + (size_t)s * F;
  float* __restrict__ od = out + (size_t)d * F;
  for (int f = lane; f < F; f += 32)
    atomicAdd(od + f, alpha * hs[f]);
}

// ------------------------- bias / relu / convert ---------------------------
__global__ void k_bias_act(const float* __restrict__ in,
                           const float* __restrict__ bias,
                           __bf16* __restrict__ outb,  // may be null
                           float*  __restrict__ outf,  // may be null
                           long long N, int F, int do_relu) {
  const long long i = blockIdx.x * (long long)blockDim.x + threadIdx.x;
  if (i >= N * F) return;
  const int f = (int)(i % F);
  float v = in[i] + bias[f];
  if (do_relu) v = fmaxf(v, 0.f);
  if (outb) outb[i] = (__bf16)v;
  if (outf) outf[i] = v;
}

// ----------------------------- final linear --------------------------------
__global__ void k_final(const float* __restrict__ h2r,
                        const float* __restrict__ wm,
                        const float* __restrict__ bm,
                        float* __restrict__ out, int N) {
  const int n = blockIdx.x * blockDim.x + threadIdx.x;
  if (n >= N) return;
  const float* __restrict__ row = h2r + (size_t)n * 16;
  float s = bm[0];
#pragma unroll
  for (int f = 0; f < 16; ++f) s += row[f] * wm[f];
  out[n] = s;
}

// ---------------------------------------------------------------------------

static inline long long cdiv(long long a, long long b) { return (a + b - 1) / b; }

extern "C" void kernel_launch(void* const* d_in, const int* in_sizes, int n_in,
                              void* d_out, int out_size, void* d_ws, size_t ws_size,
                              hipStream_t stream) {
  const float*     x   = (const float*)d_in[0];
  const long long* ei  = (const long long*)d_in[1];   // int64 [2,E]
  const float*     W1  = (const float*)d_in[2];
  const float*     a1s = (const float*)d_in[3];
  const float*     a1d = (const float*)d_in[4];
  const float*     b1  = (const float*)d_in[5];
  const float*     Wl  = (const float*)d_in[6];
  const float*     bl  = (const float*)d_in[7];
  const float*     W2  = (const float*)d_in[8];
  const float*     a2s = (const float*)d_in[9];
  const float*     a2d = (const float*)d_in[10];
  const float*     b2  = (const float*)d_in[11];
  const float*     Wm  = (const float*)d_in[12];
  const float*     bm  = (const float*)d_in[13];

  const int       Fin = 256, F1 = 128, Fl = 32, F2 = 16;
  const int       N   = in_sizes[0] / Fin;   // 50000, divisible by 16
  const long long E   = in_sizes[1] / 2;
  const long long ET  = E + N;

  // ---- workspace layout (256B aligned slices) ----
  char* wp = (char*)d_ws;
  auto alloc = [&](size_t bytes) -> void* {
    void* r = (void*)wp;
    wp += (bytes + 255) & ~(size_t)255;
    return r;
  };
  __bf16*   xb    = (__bf16*)  alloc((size_t)N * Fin * 2);
  __bf16*   w1p   = (__bf16*)  alloc((size_t)Fin * F1 * 2);
  float*    h1    = (float*)   alloc((size_t)N * F1 * 4);
  float*    as1   = (float*)   alloc((size_t)N * 4);
  float*    ad1   = (float*)   alloc((size_t)N * 4);
  unsigned* mkey  = (unsigned*)alloc((size_t)N * 4);
  float*    denom = (float*)   alloc((size_t)N * 4);
  float*    ebuf  = (float*)   alloc((size_t)ET * 4);
  float*    exbuf = (float*)   alloc((size_t)ET * 4);
  float*    agg1  = (float*)   alloc((size_t)N * F1 * 4);
  __bf16*   h1rb  = (__bf16*)  alloc((size_t)N * F1 * 2);
  __bf16*   wlp   = (__bf16*)  alloc((size_t)F1 * Fl * 2);
  float*    hl    = (float*)   alloc((size_t)N * Fl * 4);
  __bf16*   hlb   = (__bf16*)  alloc((size_t)N * Fl * 2);
  __bf16*   w2p   = (__bf16*)  alloc((size_t)Fl * F2 * 2);
  float*    h2    = (float*)   alloc((size_t)N * F2 * 4);
  float*    as2   = (float*)   alloc((size_t)N * 4);
  float*    ad2   = (float*)   alloc((size_t)N * 4);
  float*    agg2  = (float*)   alloc((size_t)N * F2 * 4);
  float*    h2r   = (float*)   alloc((size_t)N * F2 * 4);
  (void)ws_size;

  const int B = 256;
  auto g1 = [&](long long n) { return dim3((unsigned)cdiv(n, B)); };

  // ---- layer 1 GAT: h1 = x @ W1 (bf16 WMMA, f32 accum) ----
  k_cvt_bf16<<<g1(cdiv((long long)N * Fin, 4) * 1), B, 0, stream>>>(
      x, xb, (long long)N * Fin);
  k_pack_b<<<g1((long long)Fin * F1), B, 0, stream>>>(W1, w1p, F1, Fin);
  k_wmma_gemm_bf16<4><<<dim3(N / 16, F1 / 64), 32, 0, stream>>>(
      xb, w1p, h1, N, F1, Fin);

  k_alpha<<<g1((long long)N * 32), B, 0, stream>>>(h1, a1s, a1d, as1, ad1, N, F1);

  k_fill_u32<<<g1(N), B, 0, stream>>>(mkey, 0u, N);
  k_fill_f32<<<g1(N), B, 0, stream>>>(denom, 0.f, N);
  k_fill_f32<<<g1((long long)N * F1), B, 0, stream>>>(agg1, 0.f, (long long)N * F1);

  k_edge_lrelu_max<<<g1(ET), B, 0, stream>>>(ei, E, ET, as1, ad1, ebuf, mkey);
  k_edge_exp_sum  <<<g1(ET), B, 0, stream>>>(ei, E, ET, ebuf, mkey, exbuf, denom);
  k_edge_agg      <<<g1(ET * 32), B, 0, stream>>>(ei, E, ET, h1, exbuf, denom, agg1, F1);

  // relu(agg1 + b1) -> bf16 for next GEMM
  k_bias_act<<<g1((long long)N * F1), B, 0, stream>>>(agg1, b1, h1rb, nullptr,
                                                      N, F1, /*relu=*/1);

  // ---- linear 128 -> 32 : hl = h1r @ Wl + bl ----
  k_pack_b<<<g1((long long)F1 * Fl), B, 0, stream>>>(Wl, wlp, Fl, F1);
  k_wmma_gemm_bf16<2><<<dim3(N / 16, Fl / 32), 32, 0, stream>>>(
      h1rb, wlp, hl, N, Fl, F1);
  k_bias_act<<<g1((long long)N * Fl), B, 0, stream>>>(hl, bl, hlb, nullptr,
                                                      N, Fl, /*relu=*/0);

  // ---- layer 2 GAT: h2 = (hl+bl) @ W2 ----
  k_pack_b<<<g1((long long)Fl * F2), B, 0, stream>>>(W2, w2p, F2, Fl);
  k_wmma_gemm_bf16<1><<<dim3(N / 16, F2 / 16), 32, 0, stream>>>(
      hlb, w2p, h2, N, F2, Fl);

  k_alpha<<<g1((long long)N * 32), B, 0, stream>>>(h2, a2s, a2d, as2, ad2, N, F2);

  k_fill_u32<<<g1(N), B, 0, stream>>>(mkey, 0u, N);
  k_fill_f32<<<g1(N), B, 0, stream>>>(denom, 0.f, N);
  k_fill_f32<<<g1((long long)N * F2), B, 0, stream>>>(agg2, 0.f, (long long)N * F2);

  k_edge_lrelu_max<<<g1(ET), B, 0, stream>>>(ei, E, ET, as2, ad2, ebuf, mkey);
  k_edge_exp_sum  <<<g1(ET), B, 0, stream>>>(ei, E, ET, ebuf, mkey, exbuf, denom);
  k_edge_agg      <<<g1(ET * 32), B, 0, stream>>>(ei, E, ET, h2, exbuf, denom, agg2, F2);

  // relu(agg2 + b2) -> f32
  k_bias_act<<<g1((long long)N * F2), B, 0, stream>>>(agg2, b2, nullptr, h2r,
                                                      N, F2, /*relu=*/1);

  // ---- final linear 16 -> 1 ----
  k_final<<<g1(N), B, 0, stream>>>(h2r, Wm, bm, (float*)d_out, N);
}
